// encoder_33973191311955
// MI455X (gfx1250) — matrix operand order repaired and checked by reading, not compiled, not run on hardware
//
#include <hip/hip_runtime.h>
#include <hip/hip_bf16.h>

// ---------------------------------------------------------------------------
// Types for CDNA5 WMMA (wave32): v_wmma_f32_16x16x32_bf16
// ---------------------------------------------------------------------------
typedef __bf16 v16bf __attribute__((ext_vector_type(16)));
typedef __bf16 v8bf  __attribute__((ext_vector_type(8)));
typedef float  v8f   __attribute__((ext_vector_type(8)));

#define BLK_M 64
#define BLK_N 128
#define BLK_K 64
#define LDS_K 72   // padded K-stride: 144B = 36 banks -> conflict-free frags

// ---- gfx1250 async global->LDS staging (ASYNCcnt), with safe fallback ----
#if defined(__has_builtin)
#if __has_builtin(__builtin_amdgcn_global_load_async_to_lds_b128)
#define HAVE_ASYNC_LDS 1
#endif
#if __has_builtin(__builtin_amdgcn_s_wait_asynccnt)
#define HAVE_WAIT_ASYNC 1
#endif
#endif
#ifndef HAVE_ASYNC_LDS
#define HAVE_ASYNC_LDS 0
#endif
#ifndef HAVE_WAIT_ASYNC
#define HAVE_WAIT_ASYNC 0
#endif

#if HAVE_ASYNC_LDS
// Builtin signature (from hipcc diagnostic): params are generic pointers to
// 'int __attribute__((vector_size(16)))' (int4). Codegen inserts AS casts.
typedef int async_i4 __attribute__((vector_size(4 * sizeof(int))));
#define ASYNC_PTR(p) ((async_i4*)(void*)(p))
__device__ __forceinline__ void wait_async_zero() {
#if HAVE_WAIT_ASYNC
    __builtin_amdgcn_s_wait_asynccnt(0);
#else
    asm volatile("s_wait_asynccnt 0x0" ::: "memory");
#endif
}
#endif

// ---------------------------------------------------------------------------
// GEMM: C[M,N] = A[M,K] * Bw[N,K]^T (+ bias[n]),  bf16 WMMA, f32 accumulate.
// A is f32 (converted in-register) or bf16. Bw is pre-converted bf16.
// 256 threads = 8 waves (2M x 4N); wave tile 32x32 = 2x2 WMMA fragments.
// Double-buffered LDS; one barrier per K-iteration; 8 WMMAs per iteration.
// B tile (and A tile when bf16) staged with GLOBAL_LOAD_ASYNC_TO_LDS_B128.
// ---------------------------------------------------------------------------
template<typename AT, bool ADD_BIAS>
__global__ __launch_bounds__(256) void gemm_wmma_bf16(
    const AT* __restrict__ A, const __bf16* __restrict__ Bw,
    float* __restrict__ C, const float* __restrict__ bias,
    int M, int N, int K)
{
    __shared__ __align__(64) __bf16 As[2][BLK_M][LDS_K];
    __shared__ __align__(64) __bf16 Bs[2][BLK_N][LDS_K];

    constexpr bool A_BF16 = (sizeof(AT) == 2);

    const int t  = threadIdx.x;
    const int m0 = blockIdx.y * BLK_M;
    const int n0 = blockIdx.x * BLK_N;

    const int wave = t >> 5;
    const int lane = t & 31;
    const int hl   = lane >> 4;   // 16-lane half
    const int l15  = lane & 15;
    const int wm   = wave & 1;    // 2 M sub-tiles of 32
    const int wn   = wave >> 1;   // 4 N sub-tiles of 32

    // A staging: 64 rows x 64 cols, 16 elems/thread (one 16-elem chunk)
    const int ar = t >> 2;          // 0..63
    const int ac = (t & 3) << 4;    // 0,16,32,48
    // B staging: 128 rows x 64 cols, 32 elems/thread (contiguous 64B)
    const int br = t >> 1;          // 0..127
    const int bc = (t & 1) << 5;    // 0,32

    v8f acc[2][2];
#pragma unroll
    for (int mi = 0; mi < 2; ++mi)
#pragma unroll
        for (int ni = 0; ni < 2; ++ni)
            acc[mi][ni] = (v8f){0.f, 0.f, 0.f, 0.f, 0.f, 0.f, 0.f, 0.f};

    v8bf aregs[2];   // A staged per thread (register path)
    v8bf bregs[4];   // B staged per thread (fallback path only)

    // ---- register-path loaders ----
    auto loadA = [&](int k0) {
        if constexpr (!A_BF16) {
            const float* ap = (const float*)&A[(size_t)(m0 + ar) * K + k0 + ac];
            const float4 a0 = *(const float4*)(ap + 0);
            const float4 a1 = *(const float4*)(ap + 4);
            const float4 a2 = *(const float4*)(ap + 8);
            const float4 a3 = *(const float4*)(ap + 12);
            aregs[0][0] = (__bf16)a0.x; aregs[0][1] = (__bf16)a0.y;
            aregs[0][2] = (__bf16)a0.z; aregs[0][3] = (__bf16)a0.w;
            aregs[0][4] = (__bf16)a1.x; aregs[0][5] = (__bf16)a1.y;
            aregs[0][6] = (__bf16)a1.z; aregs[0][7] = (__bf16)a1.w;
            aregs[1][0] = (__bf16)a2.x; aregs[1][1] = (__bf16)a2.y;
            aregs[1][2] = (__bf16)a2.z; aregs[1][3] = (__bf16)a2.w;
            aregs[1][4] = (__bf16)a3.x; aregs[1][5] = (__bf16)a3.y;
            aregs[1][6] = (__bf16)a3.z; aregs[1][7] = (__bf16)a3.w;
        } else {
            const __bf16* ap = (const __bf16*)&A[(size_t)(m0 + ar) * K + k0 + ac];
            aregs[0] = *(const v8bf*)(ap + 0);
            aregs[1] = *(const v8bf*)(ap + 8);
        }
    };
    auto storeA = [&](int buf) {
        *(v8bf*)&As[buf][ar][ac + 0] = aregs[0];
        *(v8bf*)&As[buf][ar][ac + 8] = aregs[1];
    };
    auto loadB = [&](int k0) {
        const __bf16* bp = &Bw[(size_t)(n0 + br) * K + k0 + bc];
#pragma unroll
        for (int j = 0; j < 4; ++j)
            bregs[j] = *(const v8bf*)(bp + 8 * j);
    };
    auto storeB = [&](int buf) {
#pragma unroll
        for (int j = 0; j < 4; ++j)
            *(v8bf*)&Bs[buf][br][bc + 8 * j] = bregs[j];
    };

#if HAVE_ASYNC_LDS
    // ---- async-path stagers: global -> LDS directly, ASYNCcnt tracked ----
    auto asyncStageB = [&](int k0, int buf) {
        const __bf16* bp = &Bw[(size_t)(n0 + br) * K + k0 + bc];
        __bf16* lp = &Bs[buf][br][bc];
#pragma unroll
        for (int j = 0; j < 4; ++j)
            __builtin_amdgcn_global_load_async_to_lds_b128(
                ASYNC_PTR(bp + 8 * j), ASYNC_PTR(lp + 8 * j), 0, 0);
    };
    auto asyncStageA = [&](int k0, int buf) {   // bf16 A only (raw byte copy)
        const AT* ap = &A[(size_t)(m0 + ar) * K + k0 + ac];
        __bf16* lp = &As[buf][ar][ac];
        __builtin_amdgcn_global_load_async_to_lds_b128(
            ASYNC_PTR(ap), ASYNC_PTR(lp), 0, 0);
        __builtin_amdgcn_global_load_async_to_lds_b128(
            ASYNC_PTR(ap + 8), ASYNC_PTR(lp + 8), 0, 0);
    };
#endif

    // ---- prologue: stage tile 0 into buffer 0 ----
#if HAVE_ASYNC_LDS
    asyncStageB(0, 0);
    if constexpr (A_BF16) {
        asyncStageA(0, 0);
    } else {
        loadA(0);
        storeA(0);
    }
    wait_async_zero();
    __syncthreads();
#else
    loadA(0);
    loadB(0);
    storeA(0);
    storeB(0);
    __syncthreads();
#endif

    int cur = 0;
    for (int k0 = 0; k0 < K; k0 += BLK_K) {
        const bool more = (k0 + BLK_K) < K;
        if (more) {   // issue next tile's transfers before computing
#if HAVE_ASYNC_LDS
            asyncStageB(k0 + BLK_K, cur ^ 1);
            if constexpr (A_BF16) asyncStageA(k0 + BLK_K, cur ^ 1);
            else                  loadA(k0 + BLK_K);
#else
            loadA(k0 + BLK_K);
            loadB(k0 + BLK_K);
#endif
        }

        // ---- compute: 2 K-steps x (2x2) WMMA from LDS[cur] ----
#pragma unroll
        for (int ks = 0; ks < 2; ++ks) {
            const int kof = ks * 32;
            v16bf afr[2], bfr[2];
#pragma unroll
            for (int mi = 0; mi < 2; ++mi) {
                // A (16x32): lane half hl: e<8 -> K=e+8*hl ; e>=8 -> K=16+(e-8)+8*hl
                const __bf16* arow = &As[cur][wm * 32 + mi * 16 + l15][kof];
                v8bf lo = *(const v8bf*)(arow + 8 * hl);
                v8bf hi = *(const v8bf*)(arow + 16 + 8 * hl);
#pragma unroll
                for (int i = 0; i < 8; ++i) {
                    afr[mi][i] = lo[i]; afr[mi][i + 8] = hi[i];
                }
            }
#pragma unroll
            for (int ni = 0; ni < 2; ++ni) {
                // B (32x16): lane = N, e<8 -> K=e+8*hl ; e>=8 -> K=16+(e-8)+8*hl
                const __bf16* brow =
                    &Bs[cur][wn * 32 + ni * 16 + l15][kof + 8 * hl];
                v8bf b0 = *(const v8bf*)(brow + 0);
                v8bf b1 = *(const v8bf*)(brow + 16);
#pragma unroll
                for (int i = 0; i < 8; ++i) {
                    bfr[ni][i] = b0[i]; bfr[ni][i + 8] = b1[i];
                }
            }
#pragma unroll
            for (int mi = 0; mi < 2; ++mi)
#pragma unroll
                for (int ni = 0; ni < 2; ++ni)
                    acc[mi][ni] = __builtin_amdgcn_wmma_f32_16x16x32_bf16(
                        false, afr[mi], false, bfr[ni],
                        (short)0, acc[mi][ni], false, false);
        }

        if (more) {   // publish the other buffer: single barrier per K-iter
#if HAVE_ASYNC_LDS
            if constexpr (!A_BF16) storeA(cur ^ 1);
            wait_async_zero();        // own async copies landed in LDS
            __syncthreads();
#else
            storeA(cur ^ 1);
            storeB(cur ^ 1);
            __syncthreads();
#endif
            cur ^= 1;
        }
    }

    // ---- epilogue: C/D layout: VGPR r -> M = r + 8*hl ; lane -> N ----
#pragma unroll
    for (int ni = 0; ni < 2; ++ni) {
        const int col = n0 + wn * 32 + ni * 16 + l15;
        float bv = 0.f;
        if constexpr (ADD_BIAS) bv = bias[col];
#pragma unroll
        for (int mi = 0; mi < 2; ++mi) {
            const int rowb = m0 + wm * 32 + mi * 16 + hl * 8;
#pragma unroll
            for (int r = 0; r < 8; ++r)
                C[(size_t)(rowb + r) * N + col] = acc[mi][ni][r] + bv;
        }
    }
}

// ---------------------------------------------------------------------------
// Helpers: conversion / zero
// ---------------------------------------------------------------------------
__global__ void f32_to_bf16_kernel(const float* __restrict__ s,
                                   __bf16* __restrict__ d, int n) {
    int i = blockIdx.x * blockDim.x + threadIdx.x;
    if (i < n) d[i] = (__bf16)s[i];
}
__global__ void zero_f32_kernel(float* p, int n) {
    int i = blockIdx.x * blockDim.x + threadIdx.x;
    if (i < n) p[i] = 0.f;
}
__global__ void zero_bf16_kernel(__bf16* p, int n) {
    int i = blockIdx.x * blockDim.x + threadIdx.x;
    if (i < n) p[i] = (__bf16)0.f;
}

// ---------------------------------------------------------------------------
// BatchNorm: deterministic two-stage reduction over rows, then apply+quantize.
// ---------------------------------------------------------------------------
__global__ __launch_bounds__(256) void bn_stats1(
    const float* __restrict__ x, float* __restrict__ psum,
    float* __restrict__ psq, int rowsPerBlk, int D) {
    const int p = blockIdx.x;
    const int r0 = p * rowsPerBlk;
    for (int chBase = 0; chBase < D; chBase += 256) {
        const int ch = chBase + threadIdx.x;
        float s = 0.f, q = 0.f;
        for (int r = 0; r < rowsPerBlk; ++r) {
            float v = x[(size_t)(r0 + r) * D + ch];
            s += v; q += v * v;
        }
        psum[p * D + ch] = s;
        psq [p * D + ch] = q;
    }
}
__global__ void bn_stats2(const float* __restrict__ psum,
                          const float* __restrict__ psq,
                          float* __restrict__ mean, float* __restrict__ rvar,
                          int P, int D, float invN) {
    const int ch = blockIdx.x * blockDim.x + threadIdx.x;
    if (ch >= D) return;
    float s = 0.f, q = 0.f;
    for (int p = 0; p < P; ++p) { s += psum[p * D + ch]; q += psq[p * D + ch]; }
    const float m = s * invN;
    const float v = q * invN - m * m;
    mean[ch] = m;
    rvar[ch] = rsqrtf(v + 1e-5f);
}
__global__ void bn_apply(const float* __restrict__ x,
                         const float* __restrict__ mean,
                         const float* __restrict__ rvar,
                         const float* __restrict__ gamma,
                         const float* __restrict__ beta,
                         __bf16* __restrict__ xbn, int n, int D) {
    const int i = blockIdx.x * blockDim.x + threadIdx.x;
    if (i >= n) return;
    const int ch = i & (D - 1);
    xbn[i] = (__bf16)(gamma[ch] * (x[i] - mean[ch]) * rvar[ch] + beta[ch]);
}

// ---------------------------------------------------------------------------
// LSTM cell (per timestep): gates = h@Whh^T (from GEMM) + Gx[t] + b_ih + b_hh
// Writes h into d_out[b][t][:], c in place, and bf16 h for next-step GEMM.
// ---------------------------------------------------------------------------
__global__ void lstm_cell(const float* __restrict__ gates,
                          const float* __restrict__ Gx,
                          const float* __restrict__ bih,
                          const float* __restrict__ bhh,
                          float* __restrict__ c, float* __restrict__ out,
                          __bf16* __restrict__ hbf,
                          int t, int F, int B, int H, int T) {
    const int idx = blockIdx.x * blockDim.x + threadIdx.x;
    if (idx >= B * H) return;
    const int b  = idx / H;
    const int ch = idx - b * H;
    const float* g = &gates[(size_t)b * 4 * H];
    float pi = g[ch]         + bih[ch]         + bhh[ch];
    float pf = g[ch + H]     + bih[ch + H]     + bhh[ch + H];
    float pg = g[ch + 2 * H] + bih[ch + 2 * H] + bhh[ch + 2 * H];
    float po = g[ch + 3 * H] + bih[ch + 3 * H] + bhh[ch + 3 * H];
    if (t < F) {  // Gx rows are (b, f): row = b*F + t
        const float* gx = &Gx[(size_t)(b * F + t) * 4 * H];
        pi += gx[ch]; pf += gx[ch + H]; pg += gx[ch + 2 * H]; po += gx[ch + 3 * H];
    }
    const float si = 1.f / (1.f + expf(-pi));
    const float sf = 1.f / (1.f + expf(-pf));
    const float so = 1.f / (1.f + expf(-po));
    const float tg = tanhf(pg);
    const float cn = sf * c[idx] + si * tg;
    const float h  = so * tanhf(cn);
    c[idx] = cn;
    out[((size_t)b * T + t) * H + ch] = h;
    hbf[idx] = (__bf16)h;
}

// ---------------------------------------------------------------------------
// Host orchestration
// ---------------------------------------------------------------------------
static inline size_t align256(size_t x) { return (x + 255) & ~(size_t)255; }

extern "C" void kernel_launch(void* const* d_in, const int* in_sizes, int n_in,
                              void* d_out, int out_size, void* d_ws, size_t ws_size,
                              hipStream_t stream) {
    (void)n_in; (void)out_size; (void)ws_size; (void)in_sizes;
    constexpr int B = 128, F = 80, CAP = 20, DIN = 2048, D = 512, H = 512;
    constexpr int T = F + CAP;        // 100
    constexpr int ROWS = B * F;       // 10240

    const float* video = (const float*)d_in[0];
    // d_in[1] captions, d_in[2] max_frame: values unused (shapes are static)
    const float* W_enc = (const float*)d_in[3];
    const float* b_enc = (const float*)d_in[4];
    const float* gamma = (const float*)d_in[5];
    const float* beta  = (const float*)d_in[6];
    const float* W_ih  = (const float*)d_in[7];
    const float* W_hh  = (const float*)d_in[8];
    const float* b_ih  = (const float*)d_in[9];
    const float* b_hh  = (const float*)d_in[10];
    float* out = (float*)d_out;

    // ---- workspace layout ----
    char* ws = (char*)d_ws;
    size_t off = 0;
    float*  xbuf  = (float*) (ws + off); off = align256(off + (size_t)ROWS * D * 4);
    __bf16* xbn   = (__bf16*)(ws + off); off = align256(off + (size_t)ROWS * D * 2);
    float*  Gx    = (float*) (ws + off); off = align256(off + (size_t)ROWS * 4 * H * 4);
    float*  gates = (float*) (ws + off); off = align256(off + (size_t)B * 4 * H * 4);
    __bf16* wencB = (__bf16*)(ws + off); off = align256(off + (size_t)D * DIN * 2);
    __bf16* wihB  = (__bf16*)(ws + off); off = align256(off + (size_t)4 * H * D * 2);
    __bf16* whhB  = (__bf16*)(ws + off); off = align256(off + (size_t)4 * H * H * 2);
    float*  psum  = (float*) (ws + off); off = align256(off + (size_t)256 * D * 4);
    float*  psq   = (float*) (ws + off); off = align256(off + (size_t)256 * D * 4);
    float*  meanb = (float*) (ws + off); off = align256(off + (size_t)D * 4);
    float*  rvarb = (float*) (ws + off); off = align256(off + (size_t)D * 4);
    __bf16* hbf   = (__bf16*)(ws + off); off = align256(off + (size_t)B * H * 2);
    float*  cbuf  = (float*) (ws + off); off = align256(off + (size_t)B * H * 4);

    // ---- 0) prep: weights -> bf16, zero recurrent state ----
    f32_to_bf16_kernel<<<(D * DIN + 255) / 256, 256, 0, stream>>>(W_enc, wencB, D * DIN);
    f32_to_bf16_kernel<<<(4 * H * D + 255) / 256, 256, 0, stream>>>(W_ih, wihB, 4 * H * D);
    f32_to_bf16_kernel<<<(4 * H * H + 255) / 256, 256, 0, stream>>>(W_hh, whhB, 4 * H * H);
    zero_bf16_kernel<<<(B * H + 255) / 256, 256, 0, stream>>>(hbf, B * H);
    zero_f32_kernel <<<(B * H + 255) / 256, 256, 0, stream>>>(cbuf, B * H);

    // ---- 1) encoder GEMM: x[10240,512] = video[10240,2048] @ W_enc^T + b ----
    {
        dim3 grid(D / BLK_N, ROWS / BLK_M);
        gemm_wmma_bf16<float, true><<<grid, 256, 0, stream>>>(
            video, wencB, xbuf, b_enc, ROWS, D, DIN);
    }

    // ---- 2) BatchNorm (training stats, deterministic 2-stage) + bf16 ----
    bn_stats1<<<256, 256, 0, stream>>>(xbuf, psum, psq, ROWS / 256, D);
    bn_stats2<<<(D + 255) / 256, 256, 0, stream>>>(psum, psq, meanb, rvarb,
                                                   256, D, 1.f / (float)ROWS);
    bn_apply<<<(ROWS * D + 255) / 256, 256, 0, stream>>>(
        xbuf, meanb, rvarb, gamma, beta, xbn, ROWS * D, D);

    // ---- 3) hoisted input-gate GEMM: Gx[10240,2048] = xbn @ W_ih^T ----
    {
        dim3 grid((4 * H) / BLK_N, ROWS / BLK_M);
        gemm_wmma_bf16<__bf16, false><<<grid, 256, 0, stream>>>(
            xbn, wihB, Gx, nullptr, ROWS, 4 * H, D);
    }

    // ---- 4) recurrence: per step, gates = h @ W_hh^T then fused cell ----
    {
        dim3 gridR((4 * H) / BLK_N, B / BLK_M);
        const int cellBlocks = (B * H + 255) / 256;
        for (int t = 0; t < T; ++t) {
            gemm_wmma_bf16<__bf16, false><<<gridR, 256, 0, stream>>>(
                hbf, whhB, gates, nullptr, B, 4 * H, H);
            lstm_cell<<<cellBlocks, 256, 0, stream>>>(
                gates, Gx, b_ih, b_hh, cbuf, out, hbf, t, F, B, H, T);
        }
    }
}
